// MultiHeadAttention_884763263164
// MI455X (gfx1250) — matrix operand order repaired and checked by reading, not compiled, not run on hardware
//
#include <hip/hip_runtime.h>
#include <hip/hip_bf16.h>
#include <stddef.h>

// ---------------------------------------------------------------------------
// MHA for gfx1250 (CDNA5, wave32, WMMA f32_16x16x32_f16)
// B=2, T=2048, D=1024, H=16, DH=64
// ---------------------------------------------------------------------------

#define BB   2
#define TT   2048
#define DD   1024
#define HH   16
#define DHH  64
#define MM   (BB * TT)            // 4096 rows for the projection GEMMs

typedef __attribute__((ext_vector_type(16))) _Float16 v16h;
typedef __attribute__((ext_vector_type(8)))  float    v8f;

struct alignas(16) Half8 { _Float16 h[8]; };
union V16HU { v16h v; Half8 p8[2]; };

// --- fragment loaders (layouts per CDNA5 ISA 7.12.2, wave32) ----------------

// A-matrix 16x32 f16, row-major source with leading dim lda (elements).
// lane L: row = L%16; kb = (L<16)?0:8; elems[0..7]=K[kb..kb+7], elems[8..15]=K[kb+16..kb+23]
__device__ __forceinline__ v16h load_a_frag(const _Float16* __restrict__ A, int lda) {
    const int lane = threadIdx.x & 31;
    const int row  = lane & 15;
    const int kb   = (lane < 16) ? 0 : 8;
    const _Float16* p = A + (size_t)row * lda + kb;
    V16HU r;
    r.p8[0] = *(const Half8*)(p);
    r.p8[1] = *(const Half8*)(p + 16);
    return r.v;
}

// Same A fragment but from an f32 source (att weights), converted in-register.
__device__ __forceinline__ v16h load_a_frag_f32(const float* __restrict__ A, int lda) {
    const int lane = threadIdx.x & 31;
    const int row  = lane & 15;
    const int kb   = (lane < 16) ? 0 : 8;
    const float* p = A + (size_t)row * lda + kb;
    v16h f;
#pragma unroll
    for (int i = 0; i < 8; ++i) f[i] = (_Float16)p[i];
#pragma unroll
    for (int i = 0; i < 8; ++i) f[8 + i] = (_Float16)p[16 + i];
    return f;
}

// B-matrix 32x16 f16 from an N-major (transposed, N x K) source, leading dim ldb.
// lane L: col = L%16; kb = (L<16)?0:8; elems[0..7]=K[kb..kb+7], elems[8..15]=K[kb+16..kb+23] of column col.
__device__ __forceinline__ v16h load_b_frag(const _Float16* __restrict__ Bt, int ldb) {
    const int lane = threadIdx.x & 31;
    const int col  = lane & 15;
    const int kb   = (lane < 16) ? 0 : 8;
    const _Float16* p = Bt + (size_t)col * ldb + kb;
    V16HU r;
    r.p8[0] = *(const Half8*)(p);
    r.p8[1] = *(const Half8*)(p + 16);
    return r.v;
}

__device__ __forceinline__ v8f wmma_f16(v16h a, v16h b, v8f c) {
    return __builtin_amdgcn_wmma_f32_16x16x32_f16(false, a, false, b, (short)0, c,
                                                  false, false);
}

// --- stage 0: conversions ----------------------------------------------------

__global__ void cvt_f32_to_f16_kernel(const float* __restrict__ s,
                                      _Float16* __restrict__ d, int n) {
    for (int i = blockIdx.x * blockDim.x + threadIdx.x; i < n;
         i += gridDim.x * blockDim.x)
        d[i] = (_Float16)s[i];
}

// WT[n*K + k] = (f16) W[k*N + n]   (K = N = 1024)
__global__ void transpose_cvt_kernel(const float* __restrict__ W,
                                     _Float16* __restrict__ WT) {
    int i = blockIdx.x * 256 + threadIdx.x;   // exactly 1M threads
    int n = i >> 10;
    int k = i & 1023;
    WT[i] = (_Float16)W[(size_t)k * DD + n];
}

// --- stage 1 & 3: GEMM  C = X(f16, M x 1024) * WT^T + bias ------------------
// mode 0: write f16 head-major  [(b*H+h)*T + t]*DH + d   (used for Q with scale,
//         and K with scale=1)
// mode 2: write f16 head-transposed [(b*H+h)*DH + d]*T + t   (V)
// mode 3: write f32 row-major  m*1024 + n                    (output proj)
__global__ void gemm_qkvo_kernel(const _Float16* __restrict__ X,
                                 const _Float16* __restrict__ WT,
                                 const float* __restrict__ bias,
                                 _Float16* __restrict__ outh,
                                 float* __restrict__ outf,
                                 int mode, float scale) {
    const int wave = threadIdx.x >> 5;
    const int m0 = blockIdx.x * 64 + (wave & 1) * 32;
    const int n0 = blockIdx.y * 128 + (wave >> 1) * 32;

    v8f acc[2][2] = {};
    for (int k = 0; k < DD; k += 32) {
        v16h a0 = load_a_frag(X + (size_t)m0 * DD + k, DD);
        v16h a1 = load_a_frag(X + (size_t)(m0 + 16) * DD + k, DD);
        v16h b0 = load_b_frag(WT + (size_t)n0 * DD + k, DD);
        v16h b1 = load_b_frag(WT + (size_t)(n0 + 16) * DD + k, DD);
        acc[0][0] = wmma_f16(a0, b0, acc[0][0]);
        acc[0][1] = wmma_f16(a0, b1, acc[0][1]);
        acc[1][0] = wmma_f16(a1, b0, acc[1][0]);
        acc[1][1] = wmma_f16(a1, b1, acc[1][1]);
    }

    const int lane = threadIdx.x & 31;
    const int col  = lane & 15;
    const int rb   = (lane < 16) ? 0 : 8;
#pragma unroll
    for (int i = 0; i < 2; ++i) {
#pragma unroll
        for (int j = 0; j < 2; ++j) {
            const int n  = n0 + j * 16 + col;
            const float bv = bias[n];
#pragma unroll
            for (int r = 0; r < 8; ++r) {
                const int m = m0 + i * 16 + rb + r;
                const float v = (acc[i][j][r] + bv) * scale;
                if (mode == 3) {
                    outf[(size_t)m * DD + n] = v;
                } else {
                    const int bi = m >> 11, t = m & (TT - 1);
                    const int h  = n >> 6,  d = n & (DHH - 1);
                    size_t idx;
                    if (mode == 2)
                        idx = ((size_t)((bi * HH + h) * DHH + d)) * TT + t;
                    else
                        idx = ((size_t)((bi * HH + h) * TT + t)) * DHH + d;
                    outh[idx] = (_Float16)v;
                }
            }
        }
    }
}

// --- stage 2a: scores = q @ k^T + mask*(-1e9)  (q pre-scaled by 1/8) --------
// wave handles a 16-row x 64-col strip; K = DH = 64 -> 2 WMMA steps per tile.
__global__ void scores_kernel(const _Float16* __restrict__ qh,
                              const _Float16* __restrict__ kh,
                              const int* __restrict__ mask,
                              float* __restrict__ att) {
    const int wave = threadIdx.x >> 5;
    const int task = blockIdx.y * 8 + wave;   // 0..4095 (B*H*T/16 row tiles)
    const int cs   = blockIdx.x;              // 0..31 (64-column strips)
    const int bh = task >> 7;
    const int ti = (task & 127) << 4;
    const int b  = bh >> 4;

    const _Float16* qbase = qh + ((size_t)bh * TT + ti) * DHH;
    const _Float16* kbase = kh + (size_t)bh * TT * DHH;

    v16h a0 = load_a_frag(qbase + 0,  DHH);
    v16h a1 = load_a_frag(qbase + 32, DHH);

    v8f acc[4] = {};
#pragma unroll
    for (int j = 0; j < 4; ++j) {
        const int c0 = cs * 64 + j * 16;
        v16h b0 = load_b_frag(kbase + (size_t)c0 * DHH + 0,  DHH);
        v16h b1 = load_b_frag(kbase + (size_t)c0 * DHH + 32, DHH);
        acc[j] = wmma_f16(a0, b0, acc[j]);
        acc[j] = wmma_f16(a1, b1, acc[j]);
    }

    const int lane = threadIdx.x & 31;
    const int col  = lane & 15;
    const int rb   = (lane < 16) ? 0 : 8;
#pragma unroll
    for (int j = 0; j < 4; ++j) {
        const int jc  = cs * 64 + j * 16 + col;
        const float pen = -1e9f * (float)mask[b * TT + jc];
#pragma unroll
        for (int r = 0; r < 8; ++r) {
            const size_t ridx = (size_t)bh * TT + ti + rb + r;
            att[ridx * TT + jc] = acc[j][r] + pen;
        }
    }
}

// --- stage 2b: in-place row softmax over d_out att region -------------------
__global__ void softmax_kernel(float* __restrict__ att) {
    const int row = blockIdx.x;
    const int tid = threadIdx.x;
    float* p = att + (size_t)row * TT;

    float x[8];
#pragma unroll
    for (int i = 0; i < 8; ++i) x[i] = p[tid * 8 + i];

    __shared__ float red[256];
    float mx = x[0];
#pragma unroll
    for (int i = 1; i < 8; ++i) mx = fmaxf(mx, x[i]);
    red[tid] = mx;
    __syncthreads();
    for (int s = 128; s > 0; s >>= 1) {
        if (tid < s) red[tid] = fmaxf(red[tid], red[tid + s]);
        __syncthreads();
    }
    mx = red[0];
    __syncthreads();

    float sum = 0.f;
#pragma unroll
    for (int i = 0; i < 8; ++i) { x[i] = __expf(x[i] - mx); sum += x[i]; }
    red[tid] = sum;
    __syncthreads();
    for (int s = 128; s > 0; s >>= 1) {
        if (tid < s) red[tid] += red[tid + s];
        __syncthreads();
    }
    const float inv = 1.f / red[0];
#pragma unroll
    for (int i = 0; i < 8; ++i) p[tid * 8 + i] = x[i] * inv;
}

// --- stage 2c: out_head = P @ v  (P read f32 from d_out, V head-transposed) -
// wave handles 16 rows x full DH(64); K = T = 2048.
__global__ void pv_kernel(const float* __restrict__ att,
                          const _Float16* __restrict__ vT,
                          _Float16* __restrict__ ao) {
    const int wave = threadIdx.x >> 5;
    const int task = blockIdx.x * 8 + wave;  // 0..4095
    const int bh = task >> 7;
    const int ti = (task & 127) << 4;
    const int b  = bh >> 4, h = bh & 15;

    const float*    abase = att + ((size_t)bh * TT + ti) * TT;
    const _Float16* vb    = vT + (size_t)bh * DHH * TT;

    v8f acc[4] = {};
    for (int k = 0; k < TT; k += 32) {
        v16h a = load_a_frag_f32(abase + k, TT);
#pragma unroll
        for (int j = 0; j < 4; ++j) {
            v16h bf = load_b_frag(vb + (size_t)(j * 16) * TT + k, TT);
            acc[j] = wmma_f16(a, bf, acc[j]);
        }
    }

    const int lane = threadIdx.x & 31;
    const int col  = lane & 15;
    const int rb   = (lane < 16) ? 0 : 8;
#pragma unroll
    for (int j = 0; j < 4; ++j)
#pragma unroll
        for (int r = 0; r < 8; ++r) {
            const int t = ti + rb + r;
            ao[((size_t)(b * TT + t)) * DD + h * DHH + j * 16 + col] =
                (_Float16)acc[j][r];
        }
}

// ---------------------------------------------------------------------------

extern "C" void kernel_launch(void* const* d_in, const int* in_sizes, int n_in,
                              void* d_out, int out_size, void* d_ws, size_t ws_size,
                              hipStream_t stream) {
    const float* query  = (const float*)d_in[0];
    const float* key_in = (const float*)d_in[1];
    const float* value  = (const float*)d_in[2];
    const int*   mask   = (const int*)d_in[3];
    const float* Wq = (const float*)d_in[4];
    const float* bq = (const float*)d_in[5];
    const float* Wk = (const float*)d_in[6];
    const float* bk = (const float*)d_in[7];
    const float* Wv = (const float*)d_in[8];
    const float* bv = (const float*)d_in[9];
    const float* Wo = (const float*)d_in[10];
    const float* bo = (const float*)d_in[11];

    float* out = (float*)d_out;
    float* att = out + (size_t)BB * TT * DD;   // att_w region of d_out

    // workspace layout (f16 elements), ~64 MB total
    _Float16* ws = (_Float16*)d_ws;
    const size_t MD = (size_t)MM * DD;   // 4M
    const size_t WW = (size_t)DD * DD;   // 1M
    _Float16* xq  = ws;
    _Float16* xk  = xq + MD;
    _Float16* xv  = xk + MD;
    _Float16* wqt = xv + MD;
    _Float16* wkt = wqt + WW;
    _Float16* wvt = wkt + WW;
    _Float16* wot = wvt + WW;
    _Float16* qh  = wot + WW;
    _Float16* kh  = qh + MD;
    _Float16* vt  = kh + MD;
    _Float16* ao  = vt + MD;

    // stage 0: conversions
    cvt_f32_to_f16_kernel<<<4096, 256, 0, stream>>>(query,  xq, (int)MD);
    cvt_f32_to_f16_kernel<<<4096, 256, 0, stream>>>(key_in, xk, (int)MD);
    cvt_f32_to_f16_kernel<<<4096, 256, 0, stream>>>(value,  xv, (int)MD);
    transpose_cvt_kernel<<<4096, 256, 0, stream>>>(Wq, wqt);
    transpose_cvt_kernel<<<4096, 256, 0, stream>>>(Wk, wkt);
    transpose_cvt_kernel<<<4096, 256, 0, stream>>>(Wv, wvt);
    transpose_cvt_kernel<<<4096, 256, 0, stream>>>(Wo, wot);

    // stage 1: Q/K/V projections (Q pre-scaled by 1/sqrt(DH)=0.125)
    dim3 ggrid(MM / 64, DD / 128);
    gemm_qkvo_kernel<<<ggrid, 256, 0, stream>>>(xq, wqt, bq, qh, nullptr, 0, 0.125f);
    gemm_qkvo_kernel<<<ggrid, 256, 0, stream>>>(xk, wkt, bk, kh, nullptr, 0, 1.0f);
    gemm_qkvo_kernel<<<ggrid, 256, 0, stream>>>(xv, wvt, bv, vt, nullptr, 2, 1.0f);

    // stage 2: scores -> softmax (in place in d_out) -> P@V
    scores_kernel<<<dim3(TT / 64, (BB * HH * TT / 16) / 8), 256, 0, stream>>>(
        qh, kh, mask, att);
    softmax_kernel<<<BB * HH * TT, 256, 0, stream>>>(att);
    pv_kernel<<<(BB * HH * TT / 16) / 8, 256, 0, stream>>>(att, vt, ao);

    // stage 3: output projection (f32)
    gemm_qkvo_kernel<<<ggrid, 256, 0, stream>>>(ao, wot, bo, nullptr, out, 3, 1.0f);
}
